// HierarchicalMoEAttention_31086973288494
// MI455X (gfx1250) — matrix operand order repaired and checked by reading, compiled, not run on hardware
//
#include <hip/hip_runtime.h>
#include <hip/hip_bf16.h>
#include <math.h>

typedef _Float16 half_t;
typedef __attribute__((ext_vector_type(16))) _Float16 v16h;
typedef __attribute__((ext_vector_type(8)))  _Float16 v8h;
typedef __attribute__((ext_vector_type(8)))  float    v8f;

#define Bq 2
#define Sq 1024
#define Dq 1024
#define Hq 16
#define HDq 64
#define KDq 256
#define NEq 4
#define NSq 2
#define TOPKq 10
#define Mq (Bq*Sq)      // 2048 tokens
#define BHq (Bq*Hq)     // 32 batched heads

__device__ __forceinline__ float gelu_f(float v) {
    const float c = 0.7978845608028654f;
    return 0.5f * v * (1.f + tanhf(c * (v + 0.044715f * v * v * v)));
}

// ---------------------------------------------------------------------------
// WMMA GEMM: C[M,N] (+bias, act, scale) from f16 operands, f32 accumulate.
// LAYOUT 0 (NT): A[M,K] lda, Bt[N,K] ldb       (C = A * Bt^T)
// LAYOUT 1 (NN): A[M,K] lda, B [K,N] ldb       (C = A * B)
// LAYOUT 2 (TN): A[K,M] lda, B [K,N] ldb       (C = A^T * B)
// Workgroup tile 256x64x32, 8 wave32s, each wave computes a 32x64 C tile
// (2x4 WMMA accumulators -> 8 v_wmma per wave per K-step).
// Double-buffered LDS: next tile's global loads overlap WMMA compute;
// one workgroup barrier per K-step.
// Batched over blockIdx.z with two-level stride: z -> (zq=z/zdiv, zr=z%zdiv)
// offset = zq*Hi + zr*Lo  (addresses merged [b, s, h*64+d] layouts directly).
// Requires M%256==0, N%64==0, K%32==0 (true for every call in this model).
// act: 0=none, 1=gelu(tanh), 2=elu+1
// ---------------------------------------------------------------------------
template<int LAYOUT>
__global__ __launch_bounds__(256) void gemm_wmma(
    const half_t* __restrict__ A, const half_t* __restrict__ Bm,
    float* __restrict__ C32, half_t* __restrict__ C16,
    const float* __restrict__ bias,
    int M, int N, int K, int lda, int ldb, int ldc,
    float scale, int act, int zdiv,
    long long aHi, long long aLo, long long bHi, long long bLo,
    long long cHi, long long cLo)
{
    __shared__ half_t As[2][256][40];   // 256x32 tile per buffer, +8 half pad
    __shared__ half_t Bs[2][64][40];    // 64x32 tile per buffer (as [n][k])

    const int z  = blockIdx.z;
    const int zq = (zdiv > 1) ? z / zdiv : z;
    const int zr = (zdiv > 1) ? z % zdiv : 0;
    const half_t* Ab = A + (long long)zq * aHi + (long long)zr * aLo;
    const half_t* Bb = Bm + (long long)zq * bHi + (long long)zr * bLo;
    const long long coff = (long long)zq * cHi + (long long)zr * cLo;

    const int m0 = blockIdx.y * 256;
    const int n0 = blockIdx.x * 64;
    const int t = threadIdx.x;
    const int lane = t & 31, wid = t >> 5;          // 8 waves, wave w: rows w*32
    const int lr = lane & 15, hi = lane >> 4;

    v8f acc[2][4];
    v8f zv = {0.f,0.f,0.f,0.f,0.f,0.f,0.f,0.f};
#pragma unroll
    for (int mi = 0; mi < 2; ++mi)
#pragma unroll
        for (int ni = 0; ni < 4; ++ni) acc[mi][ni] = zv;

    v8h aReg[4];   // staged A tile (4 x v8h per thread)
    v8h bReg;      // staged B tile (1 x v8h per thread)

    const int nsteps = K >> 5;

    // ---- tile load (global -> regs) ----
    auto load_tile = [&](int k0) {
        if (LAYOUT != 2) {                                   // A[M,K]
#pragma unroll
            for (int it = 0; it < 4; ++it) {
                int v = t + it * 256;                        // 1024 x v8h
                int row = v >> 2, c8 = (v & 3) << 3;
                const half_t* gp = Ab + (long long)(m0 + row) * lda + (k0 + c8);
                if (it == 0 && k0 + 32 < K) __builtin_prefetch(gp + 32, 0, 1);
                aReg[it] = *(const v8h*)gp;
            }
        } else {                                             // A[K,M]
#pragma unroll
            for (int it = 0; it < 4; ++it) {
                int v = t + it * 256;
                int kr = v >> 5, c8 = (v & 31) << 3;         // 32 vecs per K-row
                aReg[it] = *(const v8h*)(Ab + (long long)(k0 + kr) * lda + (m0 + c8));
            }
        }
        if (LAYOUT == 0) {                                   // Bt[N,K]
            int row = t >> 2, c8 = (t & 3) << 3;
            bReg = *(const v8h*)(Bb + (long long)(n0 + row) * ldb + (k0 + c8));
        } else {                                             // B[K,N]
            int kr = t >> 3, c8 = (t & 7) << 3;
            bReg = *(const v8h*)(Bb + (long long)(k0 + kr) * ldb + (n0 + c8));
        }
    };

    // ---- tile commit (regs -> LDS buffer p) ----
    auto store_tile = [&](int p) {
        if (LAYOUT != 2) {
#pragma unroll
            for (int it = 0; it < 4; ++it) {
                int v = t + it * 256;
                int row = v >> 2, c8 = (v & 3) << 3;
                *(v8h*)&As[p][row][c8] = aReg[it];
            }
        } else {
#pragma unroll
            for (int it = 0; it < 4; ++it) {
                int v = t + it * 256;
                int kr = v >> 5, c8 = (v & 31) << 3;
#pragma unroll
                for (int e = 0; e < 8; ++e) As[p][c8 + e][kr] = aReg[it][e];
            }
        }
        if (LAYOUT == 0) {
            int row = t >> 2, c8 = (t & 3) << 3;
            *(v8h*)&Bs[p][row][c8] = bReg;
        } else {
            int kr = t >> 3, c8 = (t & 7) << 3;
#pragma unroll
            for (int e = 0; e < 8; ++e) Bs[p][c8 + e][kr] = bReg[e];
        }
    };

    // prologue: stage step 0 into buffer 0
    load_tile(0);
    store_tile(0);
    __syncthreads();

    for (int s = 0; s < nsteps; ++s) {
        const int cur = s & 1, nxt = cur ^ 1;
        const bool more = (s + 1 < nsteps);
        if (more) load_tile((s + 1) << 5);       // global loads in flight ...

        // ---- fragments per ISA 7.12.2 + 8 WMMAs, overlapping the loads ----
        v16h bfrag[4];
#pragma unroll
        for (int ni = 0; ni < 4; ++ni) {
            int bcol = ni * 16 + lr;
            const v8h b0 = *(const v8h*)&Bs[cur][bcol][hi * 8];
            const v8h b1 = *(const v8h*)&Bs[cur][bcol][16 + hi * 8];
#pragma unroll
            for (int e = 0; e < 8; ++e) { bfrag[ni][e] = b0[e]; bfrag[ni][e + 8] = b1[e]; }
        }
#pragma unroll
        for (int mi = 0; mi < 2; ++mi) {
            int arow = wid * 32 + mi * 16 + lr;
            const v8h a0 = *(const v8h*)&As[cur][arow][hi * 8];
            const v8h a1 = *(const v8h*)&As[cur][arow][16 + hi * 8];
            v16h afrag;
#pragma unroll
            for (int e = 0; e < 8; ++e) { afrag[e] = a0[e]; afrag[e + 8] = a1[e]; }
#pragma unroll
            for (int ni = 0; ni < 4; ++ni) {
                acc[mi][ni] = __builtin_amdgcn_wmma_f32_16x16x32_f16(
                    false, afrag, false, bfrag[ni], (short)0, acc[mi][ni], false, false);
            }
        }

        if (more) store_tile(nxt);               // waits on its own loads only
        __syncthreads();                         // one barrier per K-step
    }

    // ---- epilogue: scale, bias, activation, f32 or f16 store ----
#pragma unroll
    for (int mi = 0; mi < 2; ++mi) {
#pragma unroll
        for (int ni = 0; ni < 4; ++ni) {
            int col = n0 + ni * 16 + lr;
            float bv = bias ? bias[col] : 0.f;
#pragma unroll
            for (int r = 0; r < 8; ++r) {
                int row = m0 + wid * 32 + mi * 16 + hi * 8 + r;
                float vv = acc[mi][ni][r] * scale + bv;
                if (act == 1) vv = gelu_f(vv);
                else if (act == 2) vv = (vv > 0.f) ? vv + 1.f : expf(vv);
                long long off = coff + (long long)row * ldc + col;
                if (C16) C16[off] = (half_t)vv;
                else     C32[off] = vv;
            }
        }
    }
    (void)M; (void)N;
}

// ---------------------------------------------------------------------------
// Small supporting kernels
// ---------------------------------------------------------------------------
__global__ void cvt_f32_f16_kernel(const float* __restrict__ in, half_t* __restrict__ out, int n) {
    int i = blockIdx.x * 256 + threadIdx.x;
    if (i < n) out[i] = (half_t)in[i];
}

__global__ void zero_kernel(float* p, int n) {
    int i = blockIdx.x * 256 + threadIdx.x;
    if (i < n) p[i] = 0.f;
}

// transpose+convert: in f32 [R,C] -> out f16 [C,R], batched over z (stride R*C)
__global__ __launch_bounds__(256) void tcvt_kernel(const float* __restrict__ in,
                                                   half_t* __restrict__ out, int R, int C) {
    __shared__ float tile[32][33];
    size_t zo = (size_t)blockIdx.z * R * C;
    in += zo; out += zo;
    int r0 = blockIdx.y * 32, c0 = blockIdx.x * 32;
    int tc = threadIdx.x, tr = threadIdx.y;          // block (32,8)
#pragma unroll
    for (int j = 0; j < 32; j += 8) tile[tr + j][tc] = in[(size_t)(r0 + tr + j) * C + c0 + tc];
    __syncthreads();
#pragma unroll
    for (int j = 0; j < 32; j += 8)
        out[(size_t)(c0 + tr + j) * R + r0 + tc] = (half_t)tile[tc][tr + j];
}

// token importance: imp[tok] = dot(x[tok,:], sp_w) + sp_b
__global__ __launch_bounds__(256) void imp_kernel(const float* __restrict__ x,
                                                  const float* __restrict__ spw,
                                                  const float* __restrict__ spb,
                                                  float* __restrict__ imp) {
    __shared__ float sm[256];
    int row = blockIdx.x, t = threadIdx.x;
    float a = 0.f;
    for (int i = t; i < Dq; i += 256) a += x[(size_t)row * Dq + i] * spw[i];
    sm[t] = a; __syncthreads();
    for (int k = 128; k > 0; k >>= 1) { if (t < k) sm[t] += sm[t + k]; __syncthreads(); }
    if (t == 0) imp[row] = sm[0] + spb[0];
}

// top-k key mask per batch
__global__ __launch_bounds__(256) void topk_kernel(const float* __restrict__ imp,
                                                   float* __restrict__ mask) {
    __shared__ float simp[Sq];
    __shared__ float red[256];
    __shared__ int redi[256];
    int b = blockIdx.x, t = threadIdx.x;
    for (int i = t; i < Sq; i += 256) { simp[i] = imp[b * Sq + i]; mask[b * Sq + i] = 0.f; }
    __syncthreads();
    for (int r = 0; r < TOPKq; ++r) {
        float best = -3.0e38f; int bi = 0;
        for (int i = t; i < Sq; i += 256) { float v = simp[i]; if (v > best) { best = v; bi = i; } }
        red[t] = best; redi[t] = bi; __syncthreads();
        for (int k = 128; k > 0; k >>= 1) {
            if (t < k && red[t + k] > red[t]) { red[t] = red[t + k]; redi[t] = redi[t + k]; }
            __syncthreads();
        }
        if (t == 0) { mask[b * Sq + redi[0]] = 1.f; simp[redi[0]] = -3.0e38f; }
        __syncthreads();
    }
}

// masked softmax over rows of [BH*S, S]; writes f16 probs
__global__ __launch_bounds__(256) void softmax_kernel(const float* __restrict__ scores,
                                                      half_t* __restrict__ probs,
                                                      const float* __restrict__ mask) {
    __shared__ float sm[256];
    size_t row = blockIdx.x;
    int bh = (int)(row >> 10), b = bh / Hq;
    const float* src = scores + row * (size_t)Sq;
    half_t* dst = probs + row * (size_t)Sq;
    int t = threadIdx.x;
    float mx = -3.0e38f;
    for (int i = t; i < Sq; i += 256) {
        float v = src[i]; if (mask) v *= mask[b * Sq + i];
        mx = fmaxf(mx, v);
    }
    sm[t] = mx; __syncthreads();
    for (int k = 128; k > 0; k >>= 1) { if (t < k) sm[t] = fmaxf(sm[t], sm[t + k]); __syncthreads(); }
    mx = sm[0]; __syncthreads();
    float sum = 0.f;
    for (int i = t; i < Sq; i += 256) {
        float v = src[i]; if (mask) v *= mask[b * Sq + i];
        sum += expf(v - mx);
    }
    sm[t] = sum; __syncthreads();
    for (int k = 128; k > 0; k >>= 1) { if (t < k) sm[t] += sm[t + k]; __syncthreads(); }
    float inv = 1.f / sm[0];
    for (int i = t; i < Sq; i += 256) {
        float v = src[i]; if (mask) v *= mask[b * Sq + i];
        dst[i] = (half_t)(expf(v - mx) * inv);
    }
}

// layernorm(+relu/gelu) over rows of f32 [M,N] -> f16
__global__ __launch_bounds__(256) void ln_act_kernel(const float* __restrict__ in,
                                                     half_t* __restrict__ out,
                                                     const float* __restrict__ g,
                                                     const float* __restrict__ be,
                                                     int N, int act /*0 relu, 1 gelu*/) {
    __shared__ float sm[256];
    int row = blockIdx.x, t = threadIdx.x;
    const float* xr = in + (size_t)row * N;
    float s = 0.f;
    for (int i = t; i < N; i += 256) s += xr[i];
    sm[t] = s; __syncthreads();
    for (int k = 128; k > 0; k >>= 1) { if (t < k) sm[t] += sm[t + k]; __syncthreads(); }
    float mu = sm[0] / N; __syncthreads();
    float v = 0.f;
    for (int i = t; i < N; i += 256) { float d = xr[i] - mu; v += d * d; }
    sm[t] = v; __syncthreads();
    for (int k = 128; k > 0; k >>= 1) { if (t < k) sm[t] += sm[t + k]; __syncthreads(); }
    float rstd = rsqrtf(sm[0] / N + 1e-5f);
    for (int i = t; i < N; i += 256) {
        float val = (xr[i] - mu) * rstd * g[i] + be[i];
        val = (act == 0) ? fmaxf(val, 0.f) : gelu_f(val);
        out[(size_t)row * N + i] = (half_t)val;
    }
}

// level-1 router: softmax((h1 @ r1_w2) + b) over NE experts
__global__ __launch_bounds__(256) void router1_kernel(const half_t* __restrict__ h1,
                                                      const float* __restrict__ w,
                                                      const float* __restrict__ bb,
                                                      float* __restrict__ p1) {
    __shared__ float sm[256];
    int row = blockIdx.x, t = threadIdx.x;
    const half_t* hr = h1 + (size_t)row * 512;
    float logit[NEq];
    for (int e = 0; e < NEq; ++e) {
        float s = 0.f;
        for (int i = t; i < 512; i += 256) s += (float)hr[i] * w[i * NEq + e];
        sm[t] = s; __syncthreads();
        for (int k = 128; k > 0; k >>= 1) { if (t < k) sm[t] += sm[t + k]; __syncthreads(); }
        logit[e] = sm[0] + bb[e]; __syncthreads();
    }
    if (t == 0) {
        float mx = logit[0];
        for (int e = 1; e < NEq; ++e) mx = fmaxf(mx, logit[e]);
        float ex[NEq], sum = 0.f;
        for (int e = 0; e < NEq; ++e) { ex[e] = expf(logit[e] - mx); sum += ex[e]; }
        for (int e = 0; e < NEq; ++e) p1[row * NEq + e] = ex[e] / sum;
    }
}

// level-2 router: softmax((h2 @ r2_w2[i]) + b) over NS sub-experts
__global__ __launch_bounds__(256) void router2_kernel(const half_t* __restrict__ h2,
                                                      const float* __restrict__ w,
                                                      const float* __restrict__ bb,
                                                      float* __restrict__ p2) {
    __shared__ float sm[256];
    int row = blockIdx.x, t = threadIdx.x;
    const half_t* hr = h2 + (size_t)row * 256;
    float logit[NSq];
    for (int e = 0; e < NSq; ++e) {
        float s = (t < 256) ? (float)hr[t] * w[t * NSq + e] : 0.f;
        sm[t] = s; __syncthreads();
        for (int k = 128; k > 0; k >>= 1) { if (t < k) sm[t] += sm[t + k]; __syncthreads(); }
        logit[e] = sm[0] + bb[e]; __syncthreads();
    }
    if (t == 0) {
        float mx = fmaxf(logit[0], logit[1]);
        float e0 = expf(logit[0] - mx), e1 = expf(logit[1] - mx);
        float inv = 1.f / (e0 + e1);
        p2[row * NSq + 0] = e0 * inv;
        p2[row * NSq + 1] = e1 * inv;
    }
}

// depthwise temporal conv, SAME padding, 3 taps -> f16
__global__ void dwconv_kernel(const float* __restrict__ x, const float* __restrict__ w,
                              half_t* __restrict__ out) {
    int idx = blockIdx.x * 256 + threadIdx.x;   // < Mq*Dq
    int d = idx & (Dq - 1);
    int s = (idx >> 10) & (Sq - 1);
    float acc = x[idx] * w[Dq + d];
    if (s > 0)      acc += x[idx - Dq] * w[d];
    if (s < Sq - 1) acc += x[idx + Dq] * w[2 * Dq + d];
    out[idx] = (half_t)acc;
}

// column sums of k_psi: ksum[bh][kd] = sum_l kpsi[bh][l][kd]
__global__ __launch_bounds__(256) void ksum_kernel(const half_t* __restrict__ kpsi,
                                                   float* __restrict__ ksum) {
    int bh = blockIdx.x, t = threadIdx.x;       // t == kd
    const half_t* p = kpsi + (size_t)bh * Sq * KDq + t;
    float s = 0.f;
    for (int l = 0; l < Sq; ++l) s += (float)p[(size_t)l * KDq];
    ksum[bh * KDq + t] = s;
}

// den = q_phi . ksum (+eps); write attn = num/den into merged head layout (f16)
__global__ __launch_bounds__(256) void dendiv_kernel(const half_t* __restrict__ qphi,
                                                     const float* __restrict__ ksum,
                                                     const float* __restrict__ num,
                                                     half_t* __restrict__ attn) {
    __shared__ float sm[256];
    int row = blockIdx.x;                        // bh*Sq + l
    int bh = row >> 10, l = row & (Sq - 1);
    int b = bh / Hq, h = bh % Hq;
    int t = threadIdx.x;
    const half_t* qp = qphi + (size_t)bh * Sq * KDq + (size_t)l * KDq;
    sm[t] = (float)qp[t] * ksum[bh * KDq + t];
    __syncthreads();
    for (int k = 128; k > 0; k >>= 1) { if (t < k) sm[t] += sm[t + k]; __syncthreads(); }
    float den = sm[0] + 1e-8f;
    if (t < HDq) {
        size_t no = (size_t)row * HDq + t;
        attn[(size_t)b * Sq * Dq + (size_t)l * Dq + h * HDq + t] = (half_t)(num[no] / den);
    }
}

// final += p1[:,i]*p2[:,j]*s32
__global__ void accum_kernel(float* __restrict__ out, const float* __restrict__ s32,
                             const float* __restrict__ p1, const float* __restrict__ p2,
                             int ei, int sj) {
    int i = blockIdx.x * 256 + threadIdx.x;      // < Mq*Dq
    int row = i >> 10;
    out[i] += p1[row * NEq + ei] * p2[row * NSq + sj] * s32[i];
}

// ---------------------------------------------------------------------------
extern "C" void kernel_launch(void* const* d_in, const int* in_sizes, int n_in,
                              void* d_out, int out_size, void* d_ws, size_t ws_size,
                              hipStream_t stream) {
    (void)in_sizes; (void)n_in; (void)out_size; (void)ws_size;
    const float* x       = (const float*)d_in[0];
    const float* w_qkvo  = (const float*)d_in[1];
    const float* b_qkvo  = (const float*)d_in[2];
    const float* sp_w    = (const float*)d_in[3];
    const float* sp_b    = (const float*)d_in[4];
    const float* phi_w   = (const float*)d_in[5];
    const float* phi_b   = (const float*)d_in[6];
    const float* psi_w   = (const float*)d_in[7];
    const float* psi_b   = (const float*)d_in[8];
    const float* conv_dw = (const float*)d_in[9];
    const float* conv_pw = (const float*)d_in[10];
    const float* fus_w   = (const float*)d_in[11];
    const float* fus_b   = (const float*)d_in[12];
    const float* sub_w1  = (const float*)d_in[13];
    const float* sub_b1  = (const float*)d_in[14];
    const float* sub_g   = (const float*)d_in[15];
    const float* sub_be  = (const float*)d_in[16];
    const float* sub_w2  = (const float*)d_in[17];
    const float* sub_b2  = (const float*)d_in[18];
    const float* r1_w1   = (const float*)d_in[19];
    const float* r1_b1   = (const float*)d_in[20];
    const float* r1_g    = (const float*)d_in[21];
    const float* r1_be   = (const float*)d_in[22];
    const float* r1_w2   = (const float*)d_in[23];
    const float* r1_b2   = (const float*)d_in[24];
    const float* r2_w1   = (const float*)d_in[25];
    const float* r2_b1   = (const float*)d_in[26];
    const float* r2_g    = (const float*)d_in[27];
    const float* r2_be   = (const float*)d_in[28];
    const float* r2_w2   = (const float*)d_in[29];
    const float* r2_b2   = (const float*)d_in[30];
    float* out = (float*)d_out;

    // ---- workspace bump allocator ----
    size_t off = 0;
    auto alloc = [&](size_t bytes) -> char* {
        char* p = (char*)d_ws + off;
        off = (off + bytes + 255) & ~(size_t)255;
        return p;
    };
    const size_t DD = (size_t)Dq * Dq;
    half_t* xh     = (half_t*)alloc((size_t)Mq * Dq * 2);
    half_t* wqt    = (half_t*)alloc(16 * DD * 2);             // qkvo transposed
    half_t* s1t    = (half_t*)alloc(8 * DD * 2);
    half_t* s2t    = (half_t*)alloc(8 * DD * 2);
    half_t* cpt    = (half_t*)alloc(DD * 2);
    half_t* fust   = (half_t*)alloc(2 * DD * 2);              // [1024][2048]
    half_t* r1t    = (half_t*)alloc((size_t)512 * 1024 * 2);
    half_t* r2t    = (half_t*)alloc((size_t)NEq * 256 * 1024 * 2);
    half_t* phit   = (half_t*)alloc((size_t)KDq * HDq * 2);
    half_t* psit   = (half_t*)alloc((size_t)KDq * HDq * 2);
    half_t* qh     = (half_t*)alloc((size_t)Mq * Dq * 2);
    half_t* kh     = (half_t*)alloc((size_t)Mq * Dq * 2);
    half_t* vh     = (half_t*)alloc((size_t)Mq * Dq * 2);
    half_t* dwh    = (half_t*)alloc((size_t)Mq * Dq * 2);
    half_t* fusin  = (half_t*)alloc((size_t)Mq * 2 * Dq * 2); // [ao | xc]
    half_t* attnh  = (half_t*)alloc((size_t)Mq * Dq * 2);
    half_t* qphih  = (half_t*)alloc((size_t)BHq * Sq * KDq * 2);
    half_t* kpsih  = (half_t*)alloc((size_t)BHq * Sq * KDq * 2);
    half_t* kvh    = (half_t*)alloc((size_t)BHq * KDq * HDq * 2);
    half_t* eoh    = (half_t*)alloc((size_t)NEq * Mq * Dq * 2);
    half_t* h1h    = (half_t*)alloc((size_t)Mq * 512 * 2);
    half_t* h2h    = (half_t*)alloc((size_t)Mq * 256 * 2);
    half_t* sgh    = (half_t*)alloc((size_t)Mq * Dq * 2);
    half_t* probsh = (half_t*)alloc((size_t)BHq * Sq * Sq * 2);
    float*  scores = (float*)alloc((size_t)BHq * Sq * Sq * 4);
    float*  ln32   = (float*)alloc((size_t)Mq * Dq * 4);
    float*  s32    = (float*)alloc((size_t)Mq * Dq * 4);
    float*  num32  = (float*)alloc((size_t)BHq * Sq * HDq * 4);
    float*  ksum   = (float*)alloc((size_t)BHq * KDq * 4);
    float*  impb   = (float*)alloc((size_t)Mq * 4);
    float*  maskb  = (float*)alloc((size_t)Bq * Sq * 4);
    float*  p1     = (float*)alloc((size_t)Mq * NEq * 4);
    float*  p2     = (float*)alloc((size_t)Mq * NSq * 4);

    auto gemm = [&](int layout, const half_t* A, const half_t* Bm,
                    float* C32, half_t* C16, const float* bias,
                    int M, int N, int K, int lda, int ldb, int ldc,
                    float scale, int act, int Z, int zdiv,
                    long long aHi, long long aLo, long long bHi, long long bLo,
                    long long cHi, long long cLo) {
        dim3 g(N / 64, M / 256, Z), blk(256);
        if (layout == 0)
            gemm_wmma<0><<<g, blk, 0, stream>>>(A, Bm, C32, C16, bias, M, N, K, lda, ldb, ldc,
                                                scale, act, zdiv, aHi, aLo, bHi, bLo, cHi, cLo);
        else if (layout == 1)
            gemm_wmma<1><<<g, blk, 0, stream>>>(A, Bm, C32, C16, bias, M, N, K, lda, ldb, ldc,
                                                scale, act, zdiv, aHi, aLo, bHi, bLo, cHi, cLo);
        else
            gemm_wmma<2><<<g, blk, 0, stream>>>(A, Bm, C32, C16, bias, M, N, K, lda, ldb, ldc,
                                                scale, act, zdiv, aHi, aLo, bHi, bLo, cHi, cLo);
    };

    const long long SD = (long long)Sq * Dq;
    const long long SS = (long long)Sq * Sq;
    const long long SK = (long long)Sq * KDq;
    const long long KH = (long long)KDq * HDq;
    const long long SH = (long long)Sq * HDq;

    // ---- 1) precision conversion / weight transposition ----
    cvt_f32_f16_kernel<<<(Mq * Dq) / 256, 256, 0, stream>>>(x, xh, Mq * Dq);
    dim3 tb(32, 8);
    tcvt_kernel<<<dim3(Dq / 32, Dq / 32, 16), tb, 0, stream>>>(w_qkvo, wqt, Dq, Dq);
    tcvt_kernel<<<dim3(Dq / 32, Dq / 32, 8),  tb, 0, stream>>>(sub_w1, s1t, Dq, Dq);
    tcvt_kernel<<<dim3(Dq / 32, Dq / 32, 8),  tb, 0, stream>>>(sub_w2, s2t, Dq, Dq);
    tcvt_kernel<<<dim3(Dq / 32, Dq / 32, 1),  tb, 0, stream>>>(conv_pw, cpt, Dq, Dq);
    tcvt_kernel<<<dim3(Dq / 32, (2 * Dq) / 32, 1), tb, 0, stream>>>(fus_w, fust, 2 * Dq, Dq);
    tcvt_kernel<<<dim3(512 / 32, Dq / 32, 1), tb, 0, stream>>>(r1_w1, r1t, Dq, 512);
    tcvt_kernel<<<dim3(256 / 32, Dq / 32, 4), tb, 0, stream>>>(r2_w1, r2t, Dq, 256);
    tcvt_kernel<<<dim3(KDq / 32, HDq / 32, 1), tb, 0, stream>>>(phi_w, phit, HDq, KDq);
    tcvt_kernel<<<dim3(KDq / 32, HDq / 32, 1), tb, 0, stream>>>(psi_w, psit, HDq, KDq);

    // ---- 2) top-k key mask for expert 0 ----
    imp_kernel<<<Mq, 256, 0, stream>>>(x, sp_w, sp_b, impb);
    topk_kernel<<<Bq, 256, 0, stream>>>(impb, maskb);

    // ---- 3) level-1 router ----
    gemm(0, xh, r1t, ln32, nullptr, r1_b1, Mq, 512, Dq, Dq, Dq, 512,
         1.f, 0, 1, 1, 0, 0, 0, 0, 0, 0);
    ln_act_kernel<<<Mq, 256, 0, stream>>>(ln32, h1h, r1_g, r1_be, 512, 0);
    router1_kernel<<<Mq, 256, 0, stream>>>(h1h, r1_w2, r1_b2, p1);

    // ---- conv branch (expert 3 side input) ----
    dwconv_kernel<<<(Mq * Dq) / 256, 256, 0, stream>>>(x, conv_dw, dwh);
    gemm(0, dwh, cpt, nullptr, fusin + Dq, nullptr, Mq, Dq, Dq, Dq, Dq, 2 * Dq,
         1.f, 1 /*gelu*/, 1, 1, 0, 0, 0, 0, 0, 0);

    // ---- 4) experts ----
    for (int e = 0; e < NEq; ++e) {
        const half_t* Wqe = wqt + (size_t)(e * 4 + 0) * DD;
        const half_t* Wke = wqt + (size_t)(e * 4 + 1) * DD;
        const half_t* Wve = wqt + (size_t)(e * 4 + 2) * DD;
        const half_t* Woe = wqt + (size_t)(e * 4 + 3) * DD;
        const float* bqe = b_qkvo + (size_t)(e * 4 + 0) * Dq;
        const float* bke = b_qkvo + (size_t)(e * 4 + 1) * Dq;
        const float* bve = b_qkvo + (size_t)(e * 4 + 2) * Dq;
        const float* boe = b_qkvo + (size_t)(e * 4 + 3) * Dq;

        gemm(0, xh, Wqe, nullptr, qh, bqe, Mq, Dq, Dq, Dq, Dq, Dq, 1.f, 0, 1, 1, 0,0,0,0,0,0);
        gemm(0, xh, Wke, nullptr, kh, bke, Mq, Dq, Dq, Dq, Dq, Dq, 1.f, 0, 1, 1, 0,0,0,0,0,0);
        gemm(0, xh, Wve, nullptr, vh, bve, Mq, Dq, Dq, Dq, Dq, Dq, 1.f, 0, 1, 1, 0,0,0,0,0,0);

        if (e == 1) {
            // performer: phi/psi features with elu+1 epilogue
            gemm(0, qh, phit, nullptr, qphih, phi_b, Sq, KDq, HDq, Dq, HDq, KDq,
                 1.f, 2, BHq, Hq, SD, HDq, 0, 0, (long long)Hq * SK, SK);
            gemm(0, kh, psit, nullptr, kpsih, psi_b, Sq, KDq, HDq, Dq, HDq, KDq,
                 1.f, 2, BHq, Hq, SD, HDq, 0, 0, (long long)Hq * SK, SK);
            ksum_kernel<<<BHq, 256, 0, stream>>>(kpsih, ksum);
            // kv = k_psi^T @ v  (TN layout)
            gemm(2, kpsih, vh, nullptr, kvh, nullptr, KDq, HDq, Sq, KDq, Dq, HDq,
                 1.f, 0, BHq, Hq, (long long)Hq * SK, SK, SD, HDq, (long long)Hq * KH, KH);
            // num = q_phi @ kv  (NN layout)
            gemm(1, qphih, kvh, num32, nullptr, nullptr, Sq, HDq, KDq, KDq, HDq, HDq,
                 1.f, 0, BHq, Hq, (long long)Hq * SK, SK, (long long)Hq * KH, KH,
                 (long long)Hq * SH, SH);
            dendiv_kernel<<<BHq * Sq, 256, 0, stream>>>(qphih, ksum, num32, attnh);
        } else {
            // softmax attention: scores = q k^T * scale (batched over heads)
            gemm(0, qh, kh, scores, nullptr, nullptr, Sq, Sq, HDq, Dq, Dq, Sq,
                 0.125f, 0, BHq, Hq, SD, HDq, SD, HDq, (long long)Hq * SS, SS);
            softmax_kernel<<<BHq * Sq, 256, 0, stream>>>(scores, probsh,
                                                         (e == 0) ? maskb : nullptr);
            gemm(1, probsh, vh, nullptr, attnh, nullptr, Sq, HDq, Sq, Sq, Dq, Dq,
                 1.f, 0, BHq, Hq, (long long)Hq * SS, SS, SD, HDq, SD, HDq);
        }
        // output projection
        half_t* oco = (e == 3) ? fusin : (eoh + (size_t)e * Mq * Dq);
        int oldc = (e == 3) ? 2 * Dq : Dq;
        gemm(0, attnh, Woe, nullptr, oco, boe, Mq, Dq, Dq, Dq, Dq, oldc,
             1.f, 0, 1, 1, 0, 0, 0, 0, 0, 0);
    }
    // expert 3 fusion: concat([ao, xc]) @ fus_w + fus_b
    gemm(0, fusin, fust, nullptr, eoh + (size_t)3 * Mq * Dq, fus_b,
         Mq, Dq, 2 * Dq, 2 * Dq, 2 * Dq, Dq, 1.f, 0, 1, 1, 0, 0, 0, 0, 0, 0);

    // ---- 5) hierarchical mixing ----
    zero_kernel<<<(Mq * Dq) / 256, 256, 0, stream>>>(out, Mq * Dq);
    for (int i = 0; i < NEq; ++i) {
        const half_t* eo = eoh + (size_t)i * Mq * Dq;
        gemm(0, eo, r2t + (size_t)i * 256 * 1024, ln32, nullptr, r2_b1 + i * 256,
             Mq, 256, Dq, Dq, Dq, 256, 1.f, 0, 1, 1, 0, 0, 0, 0, 0, 0);
        ln_act_kernel<<<Mq, 256, 0, stream>>>(ln32, h2h, r2_g + i * 256, r2_be + i * 256, 256, 0);
        router2_kernel<<<Mq, 256, 0, stream>>>(h2h, r2_w2 + (size_t)i * 256 * NSq,
                                               r2_b2 + i * NSq, p2);
        for (int j = 0; j < NSq; ++j) {
            int sij = i * NSq + j;
            gemm(0, eo, s1t + (size_t)sij * DD, ln32, nullptr, sub_b1 + (size_t)sij * Dq,
                 Mq, Dq, Dq, Dq, Dq, Dq, 1.f, 0, 1, 1, 0, 0, 0, 0, 0, 0);
            ln_act_kernel<<<Mq, 256, 0, stream>>>(ln32, sgh, sub_g + (size_t)sij * Dq,
                                                  sub_be + (size_t)sij * Dq, Dq, 1 /*gelu*/);
            gemm(0, sgh, s2t + (size_t)sij * DD, s32, nullptr, sub_b2 + (size_t)sij * Dq,
                 Mq, Dq, Dq, Dq, Dq, Dq, 1.f, 0, 1, 1, 0, 0, 0, 0, 0, 0);
            accum_kernel<<<(Mq * Dq) / 256, 256, 0, stream>>>(out, s32, p1, p2, i, j);
        }
    }
}
